// HungarianMatcher_39384850104993
// MI455X (gfx1250) — compile-verified, matrix operand order checked by reading, not run on hardware
//
#include <hip/hip_runtime.h>

#define BSZ 8
#define Q 900
#define T 256
#define MT 2000
#define NROWS (BSZ * Q)       // 7200 = 450*16
#define NPAIRS (NROWS / 32)   // 225 row-tile pairs
#define EPSF 1e-6f

typedef __attribute__((ext_vector_type(16))) _Float16 v16h;
typedef __attribute__((ext_vector_type(8)))  _Float16 v8h;
typedef __attribute__((ext_vector_type(8)))  float    v8f;

// ---------------- pre-pass 1: sigmoid + clip -> f16 A panel [NROWS, T] ----------------
__global__ void sigmoid_to_f16(const float* __restrict__ logits,
                               _Float16* __restrict__ probs, int n) {
    int i = blockIdx.x * blockDim.x + threadIdx.x;
    if (i < n) {
        float p = 1.0f / (1.0f + __expf(-logits[i]));
        p = fminf(fmaxf(p, EPSF), 1.0f - EPSF);
        probs[i] = (_Float16)p;
    }
}

// ---------------- pre-pass 2: row-normalize labels -> f16 B panel [MT, T] ----------------
__global__ void normalize_labels_f16(const float* __restrict__ labels,
                                     _Float16* __restrict__ norm) {
    __shared__ float red[T];
    const int m = blockIdx.x;
    const int t = threadIdx.x;
    float x = labels[m * T + t];
    red[t] = x;
    __syncthreads();
    #pragma unroll
    for (int s = T / 2; s > 0; s >>= 1) {
        if (t < s) red[t] += red[t + s];
        __syncthreads();
    }
    float inv = 1.0f / (red[0] + EPSF);
    norm[m * T + t] = (_Float16)(x * inv);
}

// Fragment loader: 16-bit 16x32 A/B layout (ISA 7.12.2).
// lane<16 : K-halves {0..7} then {16..23}; lane>=16: {8..15} then {24..31}.
__device__ __forceinline__ v16h load_frag(const _Float16* __restrict__ p, int koff) {
    v8h lo = *(const v8h*)(p + koff);
    v8h hi = *(const v8h*)(p + koff + 16);
    v16h f;
    #pragma unroll
    for (int i = 0; i < 8; ++i) { f[i] = lo[i]; f[8 + i] = hi[i]; }
    return f;
}

// ---------------- fused GEMM + L1 + GIoU epilogue ----------------
__global__ __launch_bounds__(128)
void matcher_cost_wmma(const _Float16* __restrict__ Aw,   // [NROWS, T] f16 probs
                       const _Float16* __restrict__ Bw,   // [MT, T]    f16 norm labels
                       const float* __restrict__ pred_boxes,  // [NROWS, 4] cxcywh
                       const float* __restrict__ tgt_boxes,   // [MT, 4]    cxcywh
                       float* __restrict__ out) {             // [NROWS, MT]
    const int pairIdx = blockIdx.y * blockDim.y + threadIdx.y;
    if (pairIdx >= NPAIRS) return;           // whole-wave uniform exit: EXEC stays all-1 for WMMA

    const int lane  = threadIdx.x;
    const int group = lane >> 4;             // 0 or 1 (half-wave)
    const int l15   = lane & 15;
    const int rowBase0 = pairIdx * 32;
    const int rowBase1 = rowBase0 + 16;
    const int colBase  = blockIdx.x * 16;
    const int koff     = group * 8;          // half-wave K shift

    const _Float16* arow0 = Aw + (size_t)(rowBase0 + l15) * T;
    const _Float16* arow1 = Aw + (size_t)(rowBase1 + l15) * T;
    const _Float16* brow  = Bw + (size_t)(colBase  + l15) * T;

    v8f acc0 = {};
    v8f acc1 = {};
    #pragma unroll
    for (int kb = 0; kb < T; kb += 32) {
        v16h b  = load_frag(brow  + kb, koff);
        v16h a0 = load_frag(arow0 + kb, koff);
        v16h a1 = load_frag(arow1 + kb, koff);
        acc0 = __builtin_amdgcn_wmma_f32_16x16x32_f16(false, a0, false, b, (short)0, acc0, false, false);
        acc1 = __builtin_amdgcn_wmma_f32_16x16x32_f16(false, a1, false, b, (short)0, acc1, false, false);
    }

    // Epilogue: lane owns output column j = colBase + (lane&15); D row = v + 8*group.
    const int j = colBase + l15;
    const float4 tb = *(const float4*)(tgt_boxes + (size_t)j * 4);
    const float tx0 = tb.x - 0.5f * tb.z, ty0 = tb.y - 0.5f * tb.w;
    const float tx1 = tb.x + 0.5f * tb.z, ty1 = tb.y + 0.5f * tb.w;
    const float area_t = (tx1 - tx0) * (ty1 - ty0);

    #pragma unroll
    for (int h = 0; h < 2; ++h) {
        const int rb = h ? rowBase1 : rowBase0;
        const v8f acc = h ? acc1 : acc0;
        #pragma unroll
        for (int v = 0; v < 8; ++v) {
            const int i = rb + v + group * 8;
            const float4 pb = *(const float4*)(pred_boxes + (size_t)i * 4);
            // L1 on cxcywh
            const float l1 = fabsf(pb.x - tb.x) + fabsf(pb.y - tb.y) +
                             fabsf(pb.z - tb.z) + fabsf(pb.w - tb.w);
            // GIoU on xyxy
            const float px0 = pb.x - 0.5f * pb.z, py0 = pb.y - 0.5f * pb.w;
            const float px1 = pb.x + 0.5f * pb.z, py1 = pb.y + 0.5f * pb.w;
            const float area_p = (px1 - px0) * (py1 - py0);
            float iw = fminf(px1, tx1) - fmaxf(px0, tx0); iw = fmaxf(iw, 0.0f);
            float ih = fminf(py1, ty1) - fmaxf(py0, ty0); ih = fmaxf(ih, 0.0f);
            const float inter = iw * ih;
            const float uni   = area_p + area_t - inter;
            const float iou   = inter / (uni + EPSF);
            float ew = fmaxf(px1, tx1) - fminf(px0, tx0); ew = fmaxf(ew, 0.0f);
            float eh = fmaxf(py1, ty1) - fminf(py0, ty0); eh = fmaxf(eh, 0.0f);
            const float area_e = ew * eh;
            const float giou = iou - (area_e - uni) / (area_e + EPSF);
            // C = 5*L1 + 1*(-dot) + 2*(-max(giou, -1))
            const float c = 5.0f * l1 - acc[v] - 2.0f * fmaxf(giou, -1.0f);
            __builtin_nontemporal_store(c, out + (size_t)i * MT + j);  // streamed output, never re-read
        }
    }
}

extern "C" void kernel_launch(void* const* d_in, const int* in_sizes, int n_in,
                              void* d_out, int out_size, void* d_ws, size_t ws_size,
                              hipStream_t stream) {
    const float* pred_logits = (const float*)d_in[0];   // [8,900,256]
    const float* pred_boxes  = (const float*)d_in[1];   // [8,900,4]
    const float* tgt_boxes   = (const float*)d_in[2];   // [2000,4]
    const float* tgt_labels  = (const float*)d_in[3];   // [2000,256]
    float* out = (float*)d_out;                         // [8,900,2000]

    _Float16* Aw = (_Float16*)d_ws;                     // 7200*256*2 = 3.69 MB
    _Float16* Bw = Aw + (size_t)NROWS * T;              // 2000*256*2 = 1.02 MB

    const int nA = NROWS * T;
    sigmoid_to_f16<<<(nA + 255) / 256, 256, 0, stream>>>(pred_logits, Aw, nA);
    normalize_labels_f16<<<MT, T, 0, stream>>>(tgt_labels, Bw);

    dim3 block(32, 4);                                  // 4 waves / block
    dim3 grid(MT / 16, (NPAIRS + 3) / 4);               // 125 col tiles x ceil(225/4)
    matcher_cost_wmma<<<grid, block, 0, stream>>>(Aw, Bw, pred_boxes, tgt_boxes, out);
}
